// MMoE_41059887350446
// MI455X (gfx1250) — matrix operand order repaired and checked by reading, not compiled, not run on hardware
//
#include <hip/hip_runtime.h>
#include <hip/hip_bf16.h>

typedef __bf16 v16bf __attribute__((ext_vector_type(16)));
typedef float  v8f   __attribute__((ext_vector_type(8)));

#define D_MODEL 1024
#define D_FF    4096
#define N_E     8
#define N_TOK   8192   // B*N = 4*2048

// LDS strides (bf16 elements); all multiples of 8 -> 16B-aligned fragment rows
#define XSTR  1032     // 1024 + 8 pad
#define W1STR 136      // 128 + 8 pad
#define W2STR 72       // 64 + 8 pad
#define HSTR  72       // 64 + 8 pad

#define SMEM_BYTES ((64*XSTR + 64*W1STR + 256*W2STR + 64*HSTR) * 2 + 64*8*4) // 197632 B < 320KB WGP LDS

__device__ __forceinline__ ushort f2bf(float f) {
  union { float f; uint u; } v; v.f = f;
  uint r = v.u + 0x7fffu + ((v.u >> 16) & 1u);  // round-to-nearest-even
  return (ushort)(r >> 16);
}

// A fragment: 16x32 bf16 (MxK), row-major source with given stride.
// ISA layout: lane m (0-15): K 0-7 & 16-23 ; lane m+16: K 8-15 & 24-31.
__device__ __forceinline__ v16bf ld_fragA(const ushort* tile, int stride) {
  const int lane = threadIdx.x & 31;
  const int m  = lane & 15;
  const int ko = (lane >> 4) << 3;          // 0 or 8
  union { uint4 u[2]; v16bf v; } f;
  const ushort* p = tile + m * stride + ko;
  f.u[0] = *reinterpret_cast<const uint4*>(p);
  f.u[1] = *reinterpret_cast<const uint4*>(p + 16);
  return f.v;
}

// B fragment: 32x16 bf16 (KxN), sourced from an N-major (transposed) LDS tile.
// ISA layout: lane n (0-15): K 0-7 & 16-23 ; lane n+16: K 8-15 & 24-31.
__device__ __forceinline__ v16bf ld_fragB(const ushort* tileT, int stride) {
  const int lane = threadIdx.x & 31;
  const int n  = lane & 15;
  const int ko = (lane >> 4) << 3;          // 0 or 8
  union { uint4 u[2]; v16bf v; } f;
  const ushort* p = tileT + n * stride + ko;
  f.u[0] = *reinterpret_cast<const uint4*>(p);
  f.u[1] = *reinterpret_cast<const uint4*>(p + 16);
  return f.v;
}

__device__ __forceinline__ v8f wmma_bf16(v16bf a, v16bf b, v8f c) {
  return __builtin_amdgcn_wmma_f32_16x16x32_bf16(false, a, false, b, (short)0, c, false, false);
}

// ---------------- fp32 -> bf16 conversion (vectorized) ----------------
__global__ __launch_bounds__(256) void f32_to_bf16_kernel(const float* __restrict__ src,
                                                          ushort* __restrict__ dst, int n4) {
  int i = blockIdx.x * 256 + threadIdx.x;
  if (i >= n4) return;
  float4 v = reinterpret_cast<const float4*>(src)[i];
  ushort4 o;
  o.x = f2bf(v.x); o.y = f2bf(v.y); o.z = f2bf(v.z); o.w = f2bf(v.w);
  reinterpret_cast<ushort4*>(dst)[i] = o;
}

// ---------------- per-token selected-task softmax gates ----------------
__global__ __launch_bounds__(256) void gates_kernel(const float* __restrict__ x,
                                                    const int* __restrict__ b_seq,
                                                    const float* __restrict__ wg,
                                                    float* __restrict__ gates) {
  const int wv   = threadIdx.x >> 5;
  const int lane = threadIdx.x & 31;
  const int token = blockIdx.x * 8 + wv;
  if (token >= N_TOK) return;
  const int t = b_seq[token];
  if (t == 0) {                       // zero branch -> zero gates -> zero output
    if (lane < N_E) gates[token * N_E + lane] = 0.f;
    return;
  }
  const float* w  = wg + (size_t)(t - 1) * D_MODEL * N_E;
  const float* xv = x + (size_t)token * D_MODEL;
  float acc[N_E] = {0.f, 0.f, 0.f, 0.f, 0.f, 0.f, 0.f, 0.f};
  for (int d = lane; d < D_MODEL; d += 32) {
    const float xd = xv[d];
    const float4 w0 = *reinterpret_cast<const float4*>(w + d * N_E);
    const float4 w1 = *reinterpret_cast<const float4*>(w + d * N_E + 4);
    acc[0] += xd * w0.x; acc[1] += xd * w0.y; acc[2] += xd * w0.z; acc[3] += xd * w0.w;
    acc[4] += xd * w1.x; acc[5] += xd * w1.y; acc[6] += xd * w1.z; acc[7] += xd * w1.w;
  }
  #pragma unroll
  for (int e = 0; e < N_E; ++e) {
    #pragma unroll
    for (int off = 16; off > 0; off >>= 1) acc[e] += __shfl_xor(acc[e], off, 32);
  }
  if (lane == 0) {
    float mx = acc[0];
    #pragma unroll
    for (int e = 1; e < N_E; ++e) mx = fmaxf(mx, acc[e]);
    float s = 0.f, ex[N_E];
    #pragma unroll
    for (int e = 0; e < N_E; ++e) { ex[e] = __expf(acc[e] - mx); s += ex[e]; }
    const float inv = 1.f / s;
    #pragma unroll
    for (int e = 0; e < N_E; ++e) gates[token * N_E + e] = ex[e] * inv;
  }
}

// ---------------- fused MoE FFN: relu(X W1 + b1) -> gate-scale -> W2 ----------------
// Grid: (128 token tiles of 64, 4 output-col groups of 256). Block: 256 threads = 8 waves.
__global__ __launch_bounds__(256) void moe_fused_kernel(const ushort* __restrict__ xb,
                                                        const ushort* __restrict__ w1b,
                                                        const ushort* __restrict__ w2b,
                                                        const float* __restrict__ b1,
                                                        const float* __restrict__ b2,
                                                        const float* __restrict__ gates,
                                                        float* __restrict__ out) {
  extern __shared__ char smem[];
  ushort* Xs  = (ushort*)smem;                  // 64 x 1024 bf16 (stride XSTR)
  ushort* W1T = Xs + 64 * XSTR;                 // 64 f-cols x 128 k (transposed, stride W1STR)
  ushort* W2T = W1T + 64 * W1STR;               // 256 d-cols x 64 f (transposed, stride W2STR)
  ushort* Hs  = W2T + 256 * W2STR;              // 64 x 64 bf16 (stride HSTR)
  float*  Gs  = (float*)(Hs + 64 * HSTR);       // 64 x 8 gates

  const int tok0  = blockIdx.x * 64;
  const int dbase = blockIdx.y * 256;
  const int tid   = threadIdx.x;
  const int lane  = tid & 31;
  const int wv    = tid >> 5;
  const int wrow  = wv & 3;                     // C row-group (16 rows)
  const int wcol  = wv >> 2;                    // C col-half (128 cols)

  // ---- Stage X tile (64 x 1024 bf16, padded rows) via CDNA5 async global->LDS path.
  // Per-lane LDS scatter addresses preserve the XSTR padding; tracked by ASYNCcnt.
  {
    const unsigned lds_base = (unsigned)(unsigned long long)(uintptr_t)Xs; // LDS offset = addr[31:0]
    const unsigned long long gbase =
        (unsigned long long)(uintptr_t)(xb) + (unsigned long long)tok0 * (D_MODEL * 2);
    for (int u = tid; u < 64 * 128; u += 256) {      // 16B units: 128 per 2048B row
      const int row = u >> 7, c16 = u & 127;
      const unsigned lds_addr = lds_base + (unsigned)(row * (XSTR * 2) + c16 * 16);
      const unsigned long long ga = gbase + (unsigned long long)(row * 2048 + c16 * 16);
      asm volatile("global_load_async_to_lds_b128 %0, %1, off"
                   :: "v"(lds_addr), "v"(ga) : "memory");
    }
    for (int i = tid; i < 64 * N_E; i += 256) Gs[i] = gates[(size_t)tok0 * N_E + i];
    asm volatile("s_wait_asynccnt 0x0" ::: "memory");  // own async ops complete
  }
  __syncthreads();                                     // all waves' LDS data visible

  v8f Cacc[8] = {};                             // 16 rows x 128 cols per wave, persistent

  for (int e = 0; e < N_E; ++e) {
    const ushort* W1e = w1b + (size_t)e * D_MODEL * D_FF;
    const ushort* W2e = w2b + (size_t)e * D_FF * D_MODEL;
    const float*  b1e = b1 + (size_t)e * D_FF;

    for (int fc = 0; fc < D_FF / 64; ++fc) {
      const int fbase = fc << 6;
      v8f Hacc[2] = {};

      // ---- GEMM1: H[64x64] = X[64x1024] @ W1_e[1024 x fbase..+64), K staged in 128-slabs
      for (int ks = 0; ks < D_MODEL; ks += 128) {
        __syncthreads();                        // W1T reuse / prior-phase fence
        for (int i = tid; i < 4096; i += 256) { // stage W1T transposed (uint = 2 f's)
          const int kl = i >> 5, f2 = (i & 31) << 1;
          const uint v = *reinterpret_cast<const uint*>(W1e + (size_t)(ks + kl) * D_FF + fbase + f2);
          W1T[f2 * W1STR + kl]       = (ushort)(v & 0xffffu);
          W1T[(f2 + 1) * W1STR + kl] = (ushort)(v >> 16);
        }
        if (ks + 128 < D_MODEL)                 // global_prefetch_b8 next K slab
          __builtin_prefetch(W1e + (size_t)(ks + 128 + (tid >> 1)) * D_FF + fbase, 0, 0);
        __syncthreads();
        #pragma unroll
        for (int ti = 0; ti < 2; ++ti) {
          const int t = wv * 2 + ti;
          const int trow = (t >> 2) << 4, tcol = (t & 3) << 4;
          #pragma unroll
          for (int kk = 0; kk < 128; kk += 32) {
            v16bf a = ld_fragA(&Xs[trow * XSTR + ks + kk], XSTR);
            v16bf b = ld_fragB(&W1T[tcol * W1STR + kk], W1STR);
            Hacc[ti] = wmma_bf16(a, b, Hacc[ti]);
          }
        }
      }

      // ---- bias + ReLU + per-row gate scale -> bf16 H in LDS
      {
        const int n = lane & 15, half = lane >> 4;
        #pragma unroll
        for (int ti = 0; ti < 2; ++ti) {
          const int t = wv * 2 + ti;
          const int trow = (t >> 2) << 4, tcol = (t & 3) << 4;
          const float bias = b1e[fbase + tcol + n];
          #pragma unroll
          for (int r = 0; r < 8; ++r) {
            const int row = trow + r + (half << 3);
            float h = Hacc[ti][r] + bias;
            h = h > 0.f ? h : 0.f;
            h *= Gs[row * N_E + e];
            Hs[row * HSTR + tcol + n] = f2bf(h);
          }
        }
      }
      __syncthreads();

      // ---- stage W2T transposed: f in [fbase,+64) x d in [dbase,+256)
      for (int i = tid; i < 8192; i += 256) {
        const int fl = i >> 7, d2 = (i & 127) << 1;
        const uint v = *reinterpret_cast<const uint*>(W2e + (size_t)(fbase + fl) * D_MODEL + dbase + d2);
        W2T[d2 * W2STR + fl]       = (ushort)(v & 0xffffu);
        W2T[(d2 + 1) * W2STR + fl] = (ushort)(v >> 16);
      }
      __syncthreads();

      // ---- GEMM2: C += Hs[wave's 16 rows x 64] @ W2T[128 cols x 64]
      {
        v16bf a0 = ld_fragA(&Hs[(wrow << 4) * HSTR + 0], HSTR);
        v16bf a1 = ld_fragA(&Hs[(wrow << 4) * HSTR + 32], HSTR);
        #pragma unroll
        for (int ct = 0; ct < 8; ++ct) {
          const int dl = wcol * 128 + ct * 16;
          v16bf bb0 = ld_fragB(&W2T[dl * W2STR + 0], W2STR);
          v16bf bb1 = ld_fragB(&W2T[dl * W2STR + 32], W2STR);
          Cacc[ct] = wmma_bf16(a0, bb0, Cacc[ct]);
          Cacc[ct] = wmma_bf16(a1, bb1, Cacc[ct]);
        }
      }
    }
  }

  // ---- epilogue: add sum_e gate_e * b2_e, store fp32
  {
    const int n = lane & 15, half = lane >> 4;
    #pragma unroll
    for (int ct = 0; ct < 8; ++ct) {
      const int dg = dbase + wcol * 128 + ct * 16 + n;
      float bsum[8];
      #pragma unroll
      for (int e2 = 0; e2 < N_E; ++e2) bsum[e2] = b2[(size_t)e2 * D_MODEL + dg];
      #pragma unroll
      for (int r = 0; r < 8; ++r) {
        const int row = (wrow << 4) + r + (half << 3);
        float v = Cacc[ct][r];
        #pragma unroll
        for (int e2 = 0; e2 < N_E; ++e2) v += Gs[row * N_E + e2] * bsum[e2];
        out[(size_t)(tok0 + row) * D_MODEL + dg] = v;
      }
    }
  }
}

extern "C" void kernel_launch(void* const* d_in, const int* in_sizes, int n_in,
                              void* d_out, int out_size, void* d_ws, size_t ws_size,
                              hipStream_t stream) {
  (void)in_sizes; (void)n_in; (void)out_size; (void)ws_size;
  const float* x     = (const float*)d_in[0];
  const int*   b_seq = (const int*)d_in[1];
  const float* w1    = (const float*)d_in[2];
  const float* b1    = (const float*)d_in[3];
  const float* w2    = (const float*)d_in[4];
  const float* b2    = (const float*)d_in[5];
  const float* wg    = (const float*)d_in[6];
  float* out = (float*)d_out;

  char* ws = (char*)d_ws;
  size_t off = 0;
  ushort* xb  = (ushort*)(ws + off); off += (size_t)N_TOK * D_MODEL * 2;          // 16 MB
  ushort* w1b = (ushort*)(ws + off); off += (size_t)N_E * D_MODEL * D_FF * 2;     // 64 MB
  ushort* w2b = (ushort*)(ws + off); off += (size_t)N_E * D_FF * D_MODEL * 2;     // 64 MB
  float* gates = (float*)(ws + off);                                              // 256 KB

  const int nx  = N_TOK * D_MODEL / 4;           // float4 units
  const int nw  = N_E * D_MODEL * D_FF / 4;
  f32_to_bf16_kernel<<<(nx + 255) / 256, 256, 0, stream>>>(x, xb, nx);
  f32_to_bf16_kernel<<<(nw + 255) / 256, 256, 0, stream>>>(w1, w1b, nw);
  f32_to_bf16_kernel<<<(nw + 255) / 256, 256, 0, stream>>>(w2, w2b, nw);

  gates_kernel<<<N_TOK / 8, 256, 0, stream>>>(x, b_seq, wg, gates);

  dim3 grid(N_TOK / 64, D_MODEL / 256);
  moe_fused_kernel<<<grid, 256, SMEM_BYTES, stream>>>(xb, w1b, w2b, b1, b2, gates, out);
}
